// ConvLocalBlock_15083925144339
// MI455X (gfx1250) — matrix operand order repaired and checked by reading, not compiled, not run on hardware
//
#include <hip/hip_runtime.h>
#include <hip/hip_bf16.h>
#include <stddef.h>
#include <stdint.h>

// Problem constants (from reference)
#define B_    256
#define L_    512
#define CIN_  64
#define U_    256
#define FS_   5
#define L1_   508   // L - fs + 1
#define L2_   504   // L1 - fs + 1
#define KCONV 320   // fs * Cin
#define KLOC  1280  // fs * U
#define KC    32    // K chunk per LDS stage (one WMMA k-step)
#define SA_   40    // padded LDS row stride in halfs (32 + 8)
#define SB_   40
#define EPS_  0.001f

typedef __attribute__((ext_vector_type(16))) __bf16 v16bf;
typedef __attribute__((ext_vector_type(8)))  __bf16 v8bf;
typedef __attribute__((ext_vector_type(2)))  __bf16 v2bf;
typedef __attribute__((ext_vector_type(8)))  float  v8f;

__device__ __forceinline__ v8f zero_v8f() {
  v8f z;
#pragma unroll
  for (int i = 0; i < 8; ++i) z[i] = 0.0f;
  return z;
}

// Build a 16-half fragment from two 16-byte LDS chunks.
__device__ __forceinline__ v16bf frag16(const __bf16* p0, const __bf16* p1) {
  v8bf lo = *(const v8bf*)p0;
  v8bf hi = *(const v8bf*)p1;
  v16bf r;
#pragma unroll
  for (int i = 0; i < 8; ++i) { r[i] = lo[i]; r[i + 8] = hi[i]; }
  return r;
}

// ---------------------------------------------------------------------------
// Kernel 1: conv1d (valid) + bias + BN1 + ReLU  ->  y (bf16)  [B, L1, U]
// GEMM: M = B*L1 (rows are contiguous 320-f32 windows of x), K=320, N=U=256.
// Block = 256x256 output tile, 512 threads = 16 waves, each wave 64x64.
// Software-pipelined: next K-chunk prefetched into VGPRs during compute.
// ---------------------------------------------------------------------------
__global__ __launch_bounds__(512, 1)
void conv_bn_relu_kernel(const float* __restrict__ x,
                         const float* __restrict__ w,      // [320, 256] k-major
                         const float* __restrict__ cbias,  // [256]
                         const float* __restrict__ g1, const float* __restrict__ b1,
                         const float* __restrict__ m1, const float* __restrict__ v1,
                         __bf16* __restrict__ y) {
  __shared__ __bf16 sA[256 * SA_];   // 256 rows x 32 K (padded)
  __shared__ __bf16 sB[256 * SB_];   // 256 cols x 32 K (transposed, padded)

  const int t   = threadIdx.x;
  const int wid = t >> 5;
  const int lid = t & 31;
  const int l16 = lid & 15;
  const int h   = lid >> 4;          // 0: lanes 0-15, 1: lanes 16-31
  const int WM  = (wid >> 2) * 64;
  const int WN  = (wid & 3) * 64;
  const int rowBlk = blockIdx.x * 256;

  v8f acc[4][4];
#pragma unroll
  for (int i = 0; i < 4; ++i)
#pragma unroll
    for (int j = 0; j < 4; ++j) acc[i][j] = zero_v8f();

  // A loader: 2 threads per row, 16 f32 each (contiguous window of x).
  const int arow = t >> 1;
  const int akp  = (t & 1) << 4;
  const int r0   = rowBlk + arow;
  const int bb   = r0 / L1_;
  const int ll   = r0 - bb * L1_;
  const float* asrc = x + ((size_t)bb * L_ + ll) * CIN_ + akp;

  // B loader: one column n per thread, 16 consecutive k values.
  const int bn  = t & 255;
  const int bkb = (t >> 8) << 4;     // k base: 0 or 16

  // --- prologue: prefetch chunk 0 into registers ---
  float4 areg[4];
  float  breg[16];
#pragma unroll
  for (int j = 0; j < 4; ++j) areg[j] = ((const float4*)asrc)[j];
#pragma unroll
  for (int i = 0; i < 16; ++i)
    breg[i] = w[(size_t)(bkb + i) * U_ + bn];

  for (int kc = 0; kc < KCONV / KC; ++kc) {
    __syncthreads();                       // prior compute done reading LDS
    // --- commit prefetched chunk to LDS (f32 -> bf16) ---
    {
      v8bf p0, p1;
      p0[0]=(__bf16)areg[0].x; p0[1]=(__bf16)areg[0].y; p0[2]=(__bf16)areg[0].z; p0[3]=(__bf16)areg[0].w;
      p0[4]=(__bf16)areg[1].x; p0[5]=(__bf16)areg[1].y; p0[6]=(__bf16)areg[1].z; p0[7]=(__bf16)areg[1].w;
      p1[0]=(__bf16)areg[2].x; p1[1]=(__bf16)areg[2].y; p1[2]=(__bf16)areg[2].z; p1[3]=(__bf16)areg[2].w;
      p1[4]=(__bf16)areg[3].x; p1[5]=(__bf16)areg[3].y; p1[6]=(__bf16)areg[3].z; p1[7]=(__bf16)areg[3].w;
      *(v8bf*)&sA[arow * SA_ + akp]     = p0;
      *(v8bf*)&sA[arow * SA_ + akp + 8] = p1;
    }
#pragma unroll
    for (int i = 0; i < 16; i += 2) {      // pack k-pairs -> b32 LDS stores
      v2bf pk;
      pk[0] = (__bf16)breg[i];
      pk[1] = (__bf16)breg[i + 1];
      *(v2bf*)&sB[bn * SB_ + bkb + i] = pk;
    }
    // --- prefetch next chunk into registers (in flight during compute) ---
    if (kc + 1 < KCONV / KC) {
      const float* s = asrc + (kc + 1) * KC;
#pragma unroll
      for (int j = 0; j < 4; ++j) areg[j] = ((const float4*)s)[j];
#pragma unroll
      for (int i = 0; i < 16; ++i)
        breg[i] = w[(size_t)((kc + 1) * KC + bkb + i) * U_ + bn];
    }
    __syncthreads();                       // LDS chunk published

    // --- compute: 4x4 WMMA tiles per wave ---
    v16bf af[4], bfr[4];
#pragma unroll
    for (int mt = 0; mt < 4; ++mt) {
      const int row = WM + mt * 16 + l16;
      af[mt] = frag16(&sA[row * SA_ + h * 8], &sA[row * SA_ + 16 + h * 8]);
    }
#pragma unroll
    for (int nt = 0; nt < 4; ++nt) {
      const int col = WN + nt * 16 + l16;
      bfr[nt] = frag16(&sB[col * SB_ + h * 8], &sB[col * SB_ + 16 + h * 8]);
    }
#pragma unroll
    for (int mt = 0; mt < 4; ++mt)
#pragma unroll
      for (int nt = 0; nt < 4; ++nt)
        acc[mt][nt] = __builtin_amdgcn_wmma_f32_16x16x32_bf16(
            false, af[mt], false, bfr[nt], (short)0, acc[mt][nt], false, false);
  }

  // --- epilogue: bias + BN1 + ReLU, store y as bf16 ---
#pragma unroll
  for (int nt = 0; nt < 4; ++nt) {
    const int u = WN + nt * 16 + l16;
    const float cb  = cbias[u];
    const float gg  = g1[u];
    const float be  = b1[u];
    const float mu  = m1[u];
    const float inv = rsqrtf(v1[u] + EPS_);
#pragma unroll
    for (int mt = 0; mt < 4; ++mt) {
#pragma unroll
      for (int i = 0; i < 8; ++i) {
        const int rr = rowBlk + WM + mt * 16 + h * 8 + i;   // global (b,l) row
        float val = acc[mt][nt][i] + cb;
        val = gg * (val - mu) * inv + be;
        val = fmaxf(val, 0.0f);
        y[(size_t)rr * U_ + u] = (__bf16)val;
      }
    }
  }
}

// ---------------------------------------------------------------------------
// Kernel 2: locally-connected layer + bias + BN2 + ReLU -> out (f32) [B,L2,U]
// One block per position l: GEMM M=B=256, K=1280, N=U=256; weights local_w[l].
// A-tile (bf16 y -> LDS) uses GLOBAL_LOAD_ASYNC_TO_LDS_B128 (ASYNCcnt path);
// B-tile (f32 local_w -> bf16 LDS) is register-pipelined one chunk ahead and
// loaded NON-TEMPORAL: local_w (660 MB) streams once, keeping y (66 MB)
// resident in the 192 MB L2 across all 504 blocks.
// ---------------------------------------------------------------------------
__global__ __launch_bounds__(512, 1)
void local_bn_relu_kernel(const __bf16* __restrict__ y,
                          const float* __restrict__ lw,   // [L2, 1280, 256]
                          const float* __restrict__ lb,   // [L2, 256]
                          const float* __restrict__ g2, const float* __restrict__ b2,
                          const float* __restrict__ m2, const float* __restrict__ v2,
                          float* __restrict__ out) {
  __shared__ __bf16 sA[256 * SA_];
  __shared__ __bf16 sB[256 * SB_];

  const int l   = blockIdx.x;
  const int t   = threadIdx.x;
  const int wid = t >> 5;
  const int lid = t & 31;
  const int l16 = lid & 15;
  const int h   = lid >> 4;
  const int WM  = (wid >> 2) * 64;
  const int WN  = (wid & 3) * 64;

  const float* wl = lw + (size_t)l * (KLOC * U_);

  v8f acc[4][4];
#pragma unroll
  for (int i = 0; i < 4; ++i)
#pragma unroll
    for (int j = 0; j < 4; ++j) acc[i][j] = zero_v8f();

  const int arow = t >> 1;           // batch row 0..255
  const int akp  = (t & 1) << 4;
  // Per-lane LDS byte address and per-lane global byte offset for async copy.
  const unsigned ldsA  = (unsigned)(uintptr_t)&sA[arow * SA_ + akp];
  const unsigned gbase = (unsigned)(((unsigned)(arow * L1_ + l) * U_ + (unsigned)akp) * 2u);

  // B loader: one column n per thread, 16 consecutive k values.
  const int bn  = t & 255;
  const int bkb = (t >> 8) << 4;     // k base: 0 or 16

  // --- prologue: prefetch B chunk 0 into registers (non-temporal) ---
  float breg[16];
#pragma unroll
  for (int i = 0; i < 16; ++i)
    breg[i] = __builtin_nontemporal_load(&wl[(size_t)(bkb + i) * U_ + bn]);

  for (int kc = 0; kc < KLOC / KC; ++kc) {      // 40 chunks
    __syncthreads();                            // prior compute done reading LDS
    // --- A chunk: async copy bf16 y -> LDS (2 x 16B per lane, ASYNCcnt) ---
    {
      const unsigned gof = gbase + (unsigned)(kc * KC * 2);
      asm volatile(
          "global_load_async_to_lds_b128 %0, %1, %2 offset:0\n\t"
          "global_load_async_to_lds_b128 %0, %1, %2 offset:16"
          :
          : "v"(ldsA), "v"(gof), "s"(y)
          : "memory");
    }
    // --- B chunk: commit prefetched registers (f32 -> bf16, k-pair packed) ---
#pragma unroll
    for (int i = 0; i < 16; i += 2) {
      v2bf pk;
      pk[0] = (__bf16)breg[i];
      pk[1] = (__bf16)breg[i + 1];
      *(v2bf*)&sB[bn * SB_ + bkb + i] = pk;
    }
    // --- prefetch next B chunk (NT global loads in flight during compute) ---
    if (kc + 1 < KLOC / KC) {
#pragma unroll
      for (int i = 0; i < 16; ++i)
        breg[i] = __builtin_nontemporal_load(
            &wl[(size_t)((kc + 1) * KC + bkb + i) * U_ + bn]);
    }
    asm volatile("s_wait_asynccnt 0x0" ::: "memory");  // A tile landed in LDS
    __syncthreads();                                   // publish across waves

    // --- compute: 4x4 WMMA tiles per wave ---
    v16bf af[4], bfr[4];
#pragma unroll
    for (int mt = 0; mt < 4; ++mt) {
      const int row = WM + mt * 16 + l16;
      af[mt] = frag16(&sA[row * SA_ + h * 8], &sA[row * SA_ + 16 + h * 8]);
    }
#pragma unroll
    for (int nt = 0; nt < 4; ++nt) {
      const int col = WN + nt * 16 + l16;
      bfr[nt] = frag16(&sB[col * SB_ + h * 8], &sB[col * SB_ + 16 + h * 8]);
    }
#pragma unroll
    for (int mt = 0; mt < 4; ++mt)
#pragma unroll
      for (int nt = 0; nt < 4; ++nt)
        acc[mt][nt] = __builtin_amdgcn_wmma_f32_16x16x32_bf16(
            false, af[mt], false, bfr[nt], (short)0, acc[mt][nt], false, false);
  }

  // --- epilogue: local bias + BN2 + ReLU, non-temporal f32 stores ---
#pragma unroll
  for (int nt = 0; nt < 4; ++nt) {
    const int u = WN + nt * 16 + l16;
    const float cb  = lb[(size_t)l * U_ + u];
    const float gg  = g2[u];
    const float be  = b2[u];
    const float mu  = m2[u];
    const float inv = rsqrtf(v2[u] + EPS_);
#pragma unroll
    for (int mt = 0; mt < 4; ++mt) {
#pragma unroll
      for (int i = 0; i < 8; ++i) {
        const int bidx = WM + mt * 16 + h * 8 + i;  // batch index
        float val = acc[mt][nt][i] + cb;
        val = gg * (val - mu) * inv + be;
        val = fmaxf(val, 0.0f);
        __builtin_nontemporal_store(val, &out[((size_t)bidx * L2_ + l) * U_ + u]);
      }
    }
  }
}

extern "C" void kernel_launch(void* const* d_in, const int* in_sizes, int n_in,
                              void* d_out, int out_size, void* d_ws, size_t ws_size,
                              hipStream_t stream) {
  const float* x      = (const float*)d_in[0];
  const float* conv_w = (const float*)d_in[1];
  const float* conv_b = (const float*)d_in[2];
  const float* g1     = (const float*)d_in[3];
  const float* b1     = (const float*)d_in[4];
  const float* m1     = (const float*)d_in[5];
  const float* v1     = (const float*)d_in[6];
  const float* lw     = (const float*)d_in[7];
  const float* lb     = (const float*)d_in[8];
  const float* g2     = (const float*)d_in[9];
  const float* b2     = (const float*)d_in[10];
  const float* m2     = (const float*)d_in[11];
  const float* v2     = (const float*)d_in[12];

  float*  out = (float*)d_out;
  __bf16* y   = (__bf16*)d_ws;   // [B, L1, U] bf16 intermediate, ~66.5 MB

  // Stage 1: conv + BN1 + ReLU -> y (bf16)
  conv_bn_relu_kernel<<<(B_ * L1_) / 256, 512, 0, stream>>>(
      x, conv_w, conv_b, g1, b1, m1, v1, y);

  // Stage 2: locally-connected + BN2 + ReLU -> out (f32)
  local_bn_relu_kernel<<<L2_, 512, 0, stream>>>(
      y, lw, lb, g2, b2, m2, v2, out);
}
